// FlashMQAttention_43508018708729
// MI455X (gfx1250) — matrix operand order repaired and checked by reading, not compiled, not run on hardware
//
#include <hip/hip_runtime.h>

// ---------------------------------------------------------------------------
// MQA block for MI455X (gfx1250, wave32, WMMA + Tensor Data Mover):
//   pass 0: one-time f32->bf16 conversions (hidden; transposed weights)
//   pass 1: qkv GEMM (bf16 WMMA, f32 accum) -> bf16 q (pre-scaled), k, v^T
//   pass 2: causal flash MQA (WMMA QK^T / PV) -> bf16 attn
//   pass 3: proj GEMM (bf16 WMMA) + bias -> f32 out
// LDS tiles are staged by TENSOR_LOAD_TO_LDS, double-buffered: tile k+1 is
// issued before waiting on tile k (TENSORcnt completes in order, so
// s_wait_tensorcnt 2 drains exactly the current tile's two loads).
// ---------------------------------------------------------------------------

typedef __attribute__((ext_vector_type(16))) __bf16        v16bf;
typedef __attribute__((ext_vector_type(8)))  float         v8f;
typedef __attribute__((ext_vector_type(8)))  unsigned int  v8u;
typedef __attribute__((ext_vector_type(4)))  unsigned int  v4u;
typedef __attribute__((ext_vector_type(8)))  int           v8i;
typedef __attribute__((ext_vector_type(4)))  int           v4i;

__device__ __forceinline__ unsigned short f2bfu(float f) {
  // native fptrunc -> v_cvt bf16 (RNE) on gfx1250
  __bf16 h = (__bf16)f;
  return __builtin_bit_cast(unsigned short, h);
}

__device__ __forceinline__ unsigned pack2bf(float lo, float hi) {
  return (unsigned)f2bfu(lo) | ((unsigned)f2bfu(hi) << 16);
}

__device__ __forceinline__ v16bf pack_frag(const unsigned int* p,
                                           int i0, int i1, int i2, int i3,
                                           int i4, int i5, int i6, int i7) {
  v8u u;
  u[0] = p[i0]; u[1] = p[i1]; u[2] = p[i2]; u[3] = p[i3];
  u[4] = p[i4]; u[5] = p[i5]; u[6] = p[i6]; u[7] = p[i7];
  return __builtin_bit_cast(v16bf, u);
}

__device__ __forceinline__ unsigned lds_off(const void* p) {
  // low 32 bits of a flat shared pointer = LDS byte offset (ISA flat aperture)
  return (unsigned)(unsigned long long)p;
}

// Issue one 2D TDM tile load: global (bf16, row stride = stride0 elements)
// -> LDS contiguous [tile_d1][tile_d0]. D# built per ISA 8.3/8.4 bit layout.
__device__ __forceinline__ void tdm_load_2d(unsigned ldsByteOff,
                                            const void* gptr,
                                            unsigned tile_d0, unsigned tile_d1,
                                            unsigned tensor_d0, unsigned tensor_d1,
                                            unsigned long long stride0_elems) {
  const unsigned long long ga = (unsigned long long)gptr;
  v4u g0;
  g0[0] = 1u;                                              // count=1 (valid), user mode
  g0[1] = ldsByteOff;                                      // lds_addr [63:32]
  g0[2] = (unsigned)(ga & 0xFFFFFFFFu);                    // global_addr[31:0]
  g0[3] = (unsigned)((ga >> 32) & 0x01FFFFFFu)             // global_addr[56:32]
          | 0x80000000u;                                   // type=2 ("image")
  v8i g1;
  g1[0] = (int)(1u << 16);                                 // data_size=1 -> 2 bytes
  g1[1] = (int)((tensor_d0 & 0xFFFFu) << 16);              // tensor_dim0[15:0]
  g1[2] = (int)(((tensor_d0 >> 16) & 0xFFFFu) |
                ((tensor_d1 & 0xFFFFu) << 16));            // td0[31:16] | td1[15:0]
  g1[3] = (int)(((tensor_d1 >> 16) & 0xFFFFu) |
                ((tile_d0 & 0xFFFFu) << 16));              // td1[31:16] | tile_dim0
  g1[4] = (int)(tile_d1 & 0xFFFFu);                        // tile_dim1 (tile_dim2=0)
  g1[5] = (int)(unsigned)(stride0_elems & 0xFFFFFFFFu);    // dim0_stride[31:0]
  g1[6] = (int)(unsigned)((stride0_elems >> 32) & 0xFFFFu);// dim0_stride[47:32]
  g1[7] = 0;
  v4i z4 = {0, 0, 0, 0};                                   // 2D: groups 2/3 unused
  v8i z8 = {0, 0, 0, 0, 0, 0, 0, 0};
  __builtin_amdgcn_tensor_load_to_lds(g0, g1, z4, z4, z8, 0);
}

// ---------------------------------------------------------------------------
// One-time conversions.
// ---------------------------------------------------------------------------
__global__ __launch_bounds__(256) void cvt_f32_to_bf16(
    const float* __restrict__ in, unsigned short* __restrict__ out,
    long long n) {
  long long i = ((long long)blockIdx.x * 256 + threadIdx.x) * 8;
  if (i + 8 > n) return;
  const float4 a = *(const float4*)(in + i);
  const float4 b = *(const float4*)(in + i + 4);
  uint4 o;
  o.x = pack2bf(a.x, a.y);
  o.y = pack2bf(a.z, a.w);
  o.z = pack2bf(b.x, b.y);
  o.w = pack2bf(b.z, b.w);
  *(uint4*)(out + i) = o;
}

// Wt[n*K + k] = bf16(W[k*N + n])  (weights pre-transposed to [N][K])
__global__ __launch_bounds__(256) void cvt_transpose_f32_to_bf16(
    const float* __restrict__ W, unsigned short* __restrict__ Wt,
    int K, int N) {
  long long idx = (long long)blockIdx.x * 256 + threadIdx.x;
  if (idx >= (long long)N * K) return;
  int n = (int)(idx / K), k = (int)(idx % K);
  Wt[idx] = f2bfu(W[(size_t)k * N + n]);
}

// ---------------------------------------------------------------------------
// GEMM: acc[M,N] = A[M,K](bf16) @ Wt[N,K](bf16)^T, f32 accum.
// mode 0: outf = acc + bias (f32)
// mode 1: qkv split epilogue -> qg (bf16, *1/sqrt(D)), kg (bf16), vg ([B][D][S])
// Block = 256 threads (8 waves); block tile 128x128; wave tile 16x128.
// K-tile = 64 (128B TDM rows, 16 WMMAs per sync window), double-buffered.
// ---------------------------------------------------------------------------
__global__ __launch_bounds__(256) void mqa_gemm_tdm_bf16wmma(
    const unsigned short* __restrict__ A,   // [M][K] bf16
    const unsigned short* __restrict__ Wt,  // [N][K] bf16
    const float* __restrict__ bias,         // [N]
    float* __restrict__ outf,               // mode 0
    unsigned short* __restrict__ qg,        // mode 1: [T][2048]
    unsigned short* __restrict__ kg,        // mode 1: [T][128]
    unsigned short* __restrict__ vg,        // mode 1: [B][128][1024]
    int M, int N, int K, int mode) {
  constexpr float SCALE = 0.08838834764831845f;  // 1/sqrt(128)
  __shared__ __align__(16) unsigned short As[2][128][64];  // [buf][row][k] 32KB
  __shared__ __align__(16) unsigned short Bs[2][128][64];  // [buf][col][k] 32KB

  const int t    = threadIdx.x;
  const int w    = t >> 5;
  const int lane = t & 31;
  const int half = lane >> 4;
  const int l16  = lane & 15;
  const int m0   = blockIdx.y * 128;
  const int n0   = blockIdx.x * 128;

  const unsigned asOff[2] = {lds_off(&As[0][0][0]), lds_off(&As[1][0][0])};
  const unsigned bsOff[2] = {lds_off(&Bs[0][0][0]), lds_off(&Bs[1][0][0])};

  v8f acc[8] = {};

  const int NT = K / 64;
  if (w == 0) {  // prologue: tile 0 -> buffer 0
    tdm_load_2d(asOff[0], A + (size_t)m0 * K, 64, 128,
                (unsigned)K, (unsigned)M, (unsigned long long)K);
    tdm_load_2d(bsOff[0], Wt + (size_t)n0 * K, 64, 128,
                (unsigned)K, (unsigned)N, (unsigned long long)K);
  }

  for (int kt = 0; kt < NT; ++kt) {
    const int cur = kt & 1;
    __syncthreads();  // buf[cur^1] reads from iteration kt-1 are done
    if (w == 0) {
      if (kt + 1 < NT) {
        const int nk = (kt + 1) * 64;
        tdm_load_2d(asOff[cur ^ 1], A + (size_t)m0 * K + nk, 64, 128,
                    (unsigned)K, (unsigned)M, (unsigned long long)K);
        tdm_load_2d(bsOff[cur ^ 1], Wt + (size_t)n0 * K + nk, 64, 128,
                    (unsigned)K, (unsigned)N, (unsigned long long)K);
        __builtin_amdgcn_s_wait_tensorcnt(2);  // drain tile kt (in-order)
      } else {
        __builtin_amdgcn_s_wait_tensorcnt(0);
      }
    }
    __syncthreads();  // buf[cur] published to all waves

    // rows are 64 halfs = 32 uints; K-chunk c covers uints [c*16, c*16+16)
    const unsigned int* pA = (const unsigned int*)&As[cur][w * 16 + l16][0];
    const int ab = half * 4;
#pragma unroll
    for (int c = 0; c < 2; ++c) {
      const int cb = c * 16;
      v16bf af = pack_frag(pA, cb + ab + 0, cb + ab + 1, cb + ab + 2, cb + ab + 3,
                               cb + 8 + ab + 0, cb + 8 + ab + 1,
                               cb + 8 + ab + 2, cb + 8 + ab + 3);
#pragma unroll
      for (int j = 0; j < 8; ++j) {
        const unsigned int* pB = (const unsigned int*)&Bs[cur][j * 16 + l16][0];
        const int bb = cb + half * 4;
        v16bf bf = pack_frag(pB, bb + 0, bb + 1, bb + 2, bb + 3,
                                 bb + 8, bb + 9, bb + 10, bb + 11);
        acc[j] = __builtin_amdgcn_wmma_f32_16x16x32_bf16(
            false, af, false, bf, (short)0, acc[j], false, false);
      }
    }
  }

  // ---- epilogue (C layout: VGPR i -> row half*8+i, lane l16 -> col) ----
  if (mode == 0) {
#pragma unroll
    for (int j = 0; j < 8; ++j) {
      const int col = n0 + j * 16 + l16;
      const float bv = bias[col];
#pragma unroll
      for (int i = 0; i < 8; ++i) {
        const int row = m0 + w * 16 + half * 8 + i;
        outf[(size_t)row * N + col] = acc[j][i] + bv;
      }
    }
  } else if (n0 < 2048) {  // q columns: fold softmax scale, bf16
#pragma unroll
    for (int j = 0; j < 8; ++j) {
      const int col = n0 + j * 16 + l16;
      const float bv = bias[col];
#pragma unroll
      for (int i = 0; i < 8; ++i) {
        const int row = m0 + w * 16 + half * 8 + i;
        qg[(size_t)row * 2048 + col] = f2bfu((acc[j][i] + bv) * SCALE);
      }
    }
  } else if (n0 == 2048) {  // k columns 2048..2175
#pragma unroll
    for (int j = 0; j < 8; ++j) {
      const int col = n0 + j * 16 + l16;
      const int d = col - 2048;
      const float bv = bias[col];
#pragma unroll
      for (int i = 0; i < 8; ++i) {
        const int row = m0 + w * 16 + half * 8 + i;
        kg[(size_t)row * 128 + d] = f2bfu(acc[j][i] + bv);
      }
    }
  } else {  // v columns 2176..2303 -> transposed [B][128][1024]
#pragma unroll
    for (int j = 0; j < 8; ++j) {
      const int col = n0 + j * 16 + l16;
      const int d = col - 2176;
      const float bv = bias[col];
#pragma unroll
      for (int i = 0; i < 8; ++i) {
        const int row = m0 + w * 16 + half * 8 + i;
        const int b = row >> 10, s = row & 1023;
        vg[(((size_t)b * 128 + d) << 10) + s] = f2bfu(acc[j][i] + bv);
      }
    }
  }
}

// ---------------------------------------------------------------------------
// Causal flash MQA. One block per (q-tile of 128, head, batch).
// qg pre-scaled bf16 [T][2048]; kg bf16 [T][128]; vg bf16 [B][128][1024].
// Output og: bf16 [T][2048] (feeds proj GEMM directly).
// Q issued first, then double-buffered K/V tiles; in-order TENSORcnt means
// "wait <= 2" also certifies Q residency.
// ---------------------------------------------------------------------------
__global__ __launch_bounds__(256) void mqa_flash_tdm_bf16wmma(
    const unsigned short* __restrict__ qg,
    const unsigned short* __restrict__ kg,
    const unsigned short* __restrict__ vg,
    unsigned short* __restrict__ og) {
  constexpr int S = 1024, HID = 2048, D = 128;

  __shared__ __align__(16) unsigned short Qs[128][128];     // 32KB [qrow][d]
  __shared__ __align__(16) unsigned short Ks[2][32][128];   // 16KB [buf][key][d]
  __shared__ __align__(16) unsigned short Vs[2][128][32];   // 16KB [buf][d][key]
  __shared__ __align__(16) unsigned short Ps[8][16][32];    //  8KB per-wave P

  const int t    = threadIdx.x;
  const int w    = t >> 5;
  const int lane = t & 31;
  const int half = lane >> 4;
  const int l16  = lane & 15;

  const int qt = blockIdx.x;
  const int h  = blockIdx.y;
  const int b  = blockIdx.z;
  const size_t tbase = (size_t)b * S;
  const int qbase = qt * 128;

  const unsigned qsOff    = lds_off(&Qs[0][0]);
  const unsigned ksOff[2] = {lds_off(&Ks[0][0][0]), lds_off(&Ks[1][0][0])};
  const unsigned vsOff[2] = {lds_off(&Vs[0][0][0]), lds_off(&Vs[1][0][0])};

  const int ktend = (qt + 1) * 4;  // causal bound, key blocks of 32
  if (w == 0) {
    // Q tile first (completes first, in-order), then K/V tile 0 -> buffer 0
    tdm_load_2d(qsOff, qg + (tbase + qbase) * HID + h * D, 128, 128,
                (unsigned)HID, 8192u, (unsigned long long)HID);
    tdm_load_2d(ksOff[0], kg + tbase * D, 128, 32,
                (unsigned)D, 8192u, (unsigned long long)D);
    tdm_load_2d(vsOff[0], vg + ((size_t)b * D << 10), 32, 128,
                (unsigned)S, (unsigned)D, (unsigned long long)S);
  }

  float mrow[8], lrow[8];
  v8f Oacc[8] = {};
#pragma unroll
  for (int i = 0; i < 8; ++i) { mrow[i] = -3.0e38f; lrow[i] = 0.0f; }

  for (int kt = 0; kt < ktend; ++kt) {
    const int cur = kt & 1;
    __syncthreads();  // buf[cur^1] reads from iteration kt-1 are done
    if (w == 0) {
      if (kt + 1 < ktend) {
        tdm_load_2d(ksOff[cur ^ 1], kg + (tbase + (kt + 1) * 32) * D, 128, 32,
                    (unsigned)D, 8192u, (unsigned long long)D);
        tdm_load_2d(vsOff[cur ^ 1], vg + ((size_t)b * D << 10) + (kt + 1) * 32,
                    32, 128, (unsigned)S, (unsigned)D, (unsigned long long)S);
        __builtin_amdgcn_s_wait_tensorcnt(2);  // drain Q + tile kt (in-order)
      } else {
        __builtin_amdgcn_s_wait_tensorcnt(0);
      }
    }
    __syncthreads();

    // --- S = Q K^T: 16 rows x 32 keys per wave; D = 4 K-chunks of 32 ---
    v8f Sa = {}, Sb = {};
    const unsigned int* pQ = (const unsigned int*)&Qs[w * 16 + l16][0];
#pragma unroll
    for (int dc = 0; dc < 4; ++dc) {
      const int qb0 = dc * 16 + half * 4;
      v16bf qf = pack_frag(pQ, qb0 + 0, qb0 + 1, qb0 + 2, qb0 + 3,
                               qb0 + 8, qb0 + 9, qb0 + 10, qb0 + 11);
      const int kbx = dc * 16 + half * 4;
      const unsigned int* pK0 = (const unsigned int*)&Ks[cur][l16][0];
      v16bf kf0 = pack_frag(pK0, kbx + 0, kbx + 1, kbx + 2, kbx + 3,
                                 kbx + 8, kbx + 9, kbx + 10, kbx + 11);
      Sa = __builtin_amdgcn_wmma_f32_16x16x32_bf16(
          false, qf, false, kf0, (short)0, Sa, false, false);
      const unsigned int* pK1 = (const unsigned int*)&Ks[cur][16 + l16][0];
      v16bf kf1 = pack_frag(pK1, kbx + 0, kbx + 1, kbx + 2, kbx + 3,
                                 kbx + 8, kbx + 9, kbx + 10, kbx + 11);
      Sb = __builtin_amdgcn_wmma_f32_16x16x32_bf16(
          false, qf, false, kf1, (short)0, Sb, false, false);
    }

    // --- causal mask ---
    const int rowg0 = qbase + w * 16 + half * 8;
    const int colga = kt * 32 + l16;
    const int colgb = colga + 16;
#pragma unroll
    for (int i = 0; i < 8; ++i) {
      const int rg = rowg0 + i;
      if (colga > rg) Sa[i] = -3.0e38f;
      if (colgb > rg) Sb[i] = -3.0e38f;
    }

    // --- online softmax (row reduce across 16 lanes of each half) ---
#pragma unroll
    for (int i = 0; i < 8; ++i) {
      float mx = fmaxf(Sa[i], Sb[i]);
      mx = fmaxf(mx, __shfl_xor(mx, 1, 32));
      mx = fmaxf(mx, __shfl_xor(mx, 2, 32));
      mx = fmaxf(mx, __shfl_xor(mx, 4, 32));
      mx = fmaxf(mx, __shfl_xor(mx, 8, 32));
      const float mnew = fmaxf(mrow[i], mx);
      const float corr = __expf(mrow[i] - mnew);
      const float pa = __expf(Sa[i] - mnew);
      const float pb = __expf(Sb[i] - mnew);
      float ls = pa + pb;
      ls += __shfl_xor(ls, 1, 32);
      ls += __shfl_xor(ls, 2, 32);
      ls += __shfl_xor(ls, 4, 32);
      ls += __shfl_xor(ls, 8, 32);
      lrow[i] = lrow[i] * corr + ls;
      mrow[i] = mnew;
#pragma unroll
      for (int j = 0; j < 8; ++j) Oacc[j][i] *= corr;
      Sa[i] = pa;
      Sb[i] = pb;
    }

    // --- C-layout -> A-layout for P via wave-private LDS scratch ---
#pragma unroll
    for (int i = 0; i < 8; ++i) {
      Ps[w][half * 8 + i][l16]      = f2bfu(Sa[i]);
      Ps[w][half * 8 + i][16 + l16] = f2bfu(Sb[i]);
    }
    asm volatile("s_wait_dscnt 0" ::: "memory");  // wave-private ordering

    const unsigned int* pP = (const unsigned int*)&Ps[w][l16][0];
    const int pb0 = half * 4;
    v16bf pf = pack_frag(pP, pb0 + 0, pb0 + 1, pb0 + 2, pb0 + 3,
                             pb0 + 8, pb0 + 9, pb0 + 10, pb0 + 11);

    // --- O += P V ---
#pragma unroll
    for (int j = 0; j < 8; ++j) {
      const unsigned int* pV = (const unsigned int*)&Vs[cur][j * 16 + l16][0];
      const int vb = half * 4;
      v16bf vf = pack_frag(pV, vb + 0, vb + 1, vb + 2, vb + 3,
                               vb + 8, vb + 9, vb + 10, vb + 11);
      Oacc[j] = __builtin_amdgcn_wmma_f32_16x16x32_bf16(
          false, pf, false, vf, (short)0, Oacc[j], false, false);
    }
  }

  // --- normalize + bf16 store (feeds proj GEMM A operand) ---
#pragma unroll
  for (int i = 0; i < 8; ++i) {
    const float inv = 1.0f / lrow[i];
    const int row = qbase + w * 16 + half * 8 + i;
    unsigned short* dst = og + (tbase + row) * HID + h * D;
#pragma unroll
    for (int j = 0; j < 8; ++j) dst[j * 16 + l16] = f2bfu(Oacc[j][i] * inv);
  }
}

// ---------------------------------------------------------------------------
extern "C" void kernel_launch(void* const* d_in, const int* in_sizes, int n_in,
                              void* d_out, int out_size, void* d_ws, size_t ws_size,
                              hipStream_t stream) {
  (void)in_sizes; (void)n_in; (void)out_size; (void)ws_size;
  const float* hidden = (const float*)d_in[0];   // [8192, 2048]
  const float* w_attn = (const float*)d_in[1];   // [2048, 2304]
  const float* b_attn = (const float*)d_in[2];   // [2304]
  const float* w_proj = (const float*)d_in[3];   // [2048, 2048]
  const float* b_proj = (const float*)d_in[4];   // [2048]
  float* out = (float*)d_out;                    // [8192, 2048]

  constexpr int M = 8192, K = 2048, NQKV = 2304, NP = 2048;
  constexpr int B = 8, S = 1024, D = 128;

  unsigned short* p   = (unsigned short*)d_ws;
  unsigned short* hb  = p; p += (size_t)M * K;        // hidden bf16 [M][K]
  unsigned short* wat = p; p += (size_t)NQKV * K;     // w_attn^T bf16 [NQKV][K]
  unsigned short* wpt = p; p += (size_t)NP * K;       // w_proj^T bf16 [NP][K]
  unsigned short* qg  = p; p += (size_t)M * 2048;     // q bf16 (pre-scaled)
  unsigned short* kg  = p; p += (size_t)M * D;        // k bf16 [T][128]
  unsigned short* vg  = p; p += (size_t)B * D * S;    // v bf16 [B][128][1024]
  unsigned short* ag  = p; p += (size_t)M * 2048;     // attn bf16 [T][2048]

  dim3 blk(256);
  cvt_f32_to_bf16<<<(unsigned)(((size_t)M * K / 8 + 255) / 256), blk, 0, stream>>>(
      hidden, hb, (long long)M * K);
  cvt_transpose_f32_to_bf16<<<(unsigned)(((size_t)NQKV * K + 255) / 256), blk, 0, stream>>>(
      w_attn, wat, K, NQKV);
  cvt_transpose_f32_to_bf16<<<(unsigned)(((size_t)NP * K + 255) / 256), blk, 0, stream>>>(
      w_proj, wpt, K, NP);

  mqa_gemm_tdm_bf16wmma<<<dim3(NQKV / 128, M / 128), blk, 0, stream>>>(
      hb, wat, b_attn, nullptr, qg, kg, vg, M, NQKV, K, /*mode=*/1);

  mqa_flash_tdm_bf16wmma<<<dim3(8, 16, 8), blk, 0, stream>>>(qg, kg, vg, ag);

  mqa_gemm_tdm_bf16wmma<<<dim3(NP / 128, M / 128), blk, 0, stream>>>(
      ag, wpt, b_proj, out, nullptr, nullptr, nullptr, M, NP, K, /*mode=*/0);
}
